// TwistorTitansRecurrent_19920058319401
// MI455X (gfx1250) — compile-verified, matrix operand order checked by reading, not run on hardware
//
#include <hip/hip_runtime.h>

// ---------------------------------------------------------------------------
// TwistorTitansRecurrent — CDNA5 (gfx1250) WMMA + TDM implementation
//   B=16, S=2048, D=1024, H=1024, HALF=512, CHUNK=512, NC=4, CYCLES=3
// ---------------------------------------------------------------------------

#define B_      16
#define S_      2048
#define D_      1024
#define H_      1024
#define HALF_   512
#define CHUNK_  512
#define NC_     4
#define NTOK_   32768   // B*S
#define COUPLE_ 0.1f

typedef __attribute__((ext_vector_type(16))) _Float16 v16h;
typedef __attribute__((ext_vector_type(8)))  _Float16 v8h;
typedef __attribute__((ext_vector_type(8)))  float    v8f;
typedef __attribute__((ext_vector_type(4)))  unsigned u32x4;
typedef __attribute__((ext_vector_type(8)))  int      i32x8;
typedef __attribute__((ext_vector_type(4)))  int      i32x4;

union Frag16 { v16h v; v8h h[2]; };

// ---------------------------------------------------------------------------
// TDM: 2D tile load (16 rows x 512 f16 elements, row stride 0x40000 elements)
// from global into LDS at byte offset lds_addr. D# per CDNA5 ISA.
// ---------------------------------------------------------------------------
__device__ __forceinline__ void tdm_load_tile(unsigned lds_addr,
                                              const _Float16* gptr) {
    unsigned long long ga = (unsigned long long)(uintptr_t)gptr;
    u32x4 g0;
    g0[0] = 1u;                                   // count=1, user descriptor
    g0[1] = lds_addr;                             // LDS byte address
    g0[2] = (unsigned)(ga & 0xffffffffu);         // global_addr[31:0]
    g0[3] = (unsigned)((ga >> 32) & 0x01ffffffu)  // global_addr[56:32]
          | 0x80000000u;                          // type = 2 ("image")
    i32x8 g1;
    g1[0] = 0x00010000;   // workgroup_mask=0, data_size=1 (2 bytes)
    g1[1] = 0;            // atomic_barrier_address=0, tensor_dim0 lo16 = 0
    g1[2] = 0x00100004;   // tensor_dim0 hi = 4 (0x40000), tensor_dim1 = 16
    g1[3] = 0x02000000;   // tensor_dim1 hi = 0, tile_dim0 = 512
    g1[4] = 16;           // tile_dim1 = 16, tile_dim2 = 0
    g1[5] = 0x40000;      // tensor_dim0_stride = 262144 elements
    g1[6] = 0;            // stride hi, tensor_dim1_stride lo
    g1[7] = 0;
    i32x4 z4 = {0, 0, 0, 0};
#if defined(__clang_major__) && (__clang_major__ >= 23)
    i32x8 z8 = {0, 0, 0, 0, 0, 0, 0, 0};
    __builtin_amdgcn_tensor_load_to_lds(g0, g1, z4, z4, z8, 0);
#else
    __builtin_amdgcn_tensor_load_to_lds(g0, g1, z4, z4, 0);
#endif
}

// ---------------------------------------------------------------------------
// Pack a row-major f32 weight (Krows x Ncols) into WMMA B-fragment layout:
//   tile = nt*ktiles + kt ; each lane owns 16 contiguous halves.
//   element(lane, i) = W[kt*32 + kof][nt*16 + (lane&15)]
//   kof = (lane>=16 ? 8 : 0) + (i>=8 ? 16 : 0) + (i&7)     (ISA 7.12.2)
// ---------------------------------------------------------------------------
__global__ void pack_w_kernel(const float* __restrict__ W,
                              _Float16* __restrict__ P,
                              int Krows, int Ncols) {
    const int ktiles = Krows >> 5;
    const int tile   = blockIdx.x;          // nt*ktiles + kt
    const int nt     = tile / ktiles;
    const int kt     = tile % ktiles;
    const int lane   = threadIdx.x;         // 0..31
    const int col    = nt * 16 + (lane & 15);
    const int kb     = kt * 32 + ((lane >> 4) << 3);

    Frag16 f;
    #pragma unroll
    for (int i = 0; i < 16; ++i) {
        int k = kb + ((i >> 3) << 4) + (i & 7);
        f.v[i] = (_Float16)W[(size_t)k * Ncols + col];
    }
    _Float16* dst = P + (size_t)tile * 512 + lane * 16;
    *(v8h*)dst       = f.h[0];
    *(v8h*)(dst + 8) = f.h[1];
}

// ---------------------------------------------------------------------------
// Projection GEMM: Y(32768 x 1536) = X(32768 x 1024) @ [Wxo|Wxp|Wa] + bias,
// sigmoid on the alpha third. Stores f16 row-major (32768 x 512) per output.
// Block: 256 threads (8 waves). Wave w -> mtile = blockIdx.x*8 + w.
// blockIdx.y picks 8 N-tiles; region (xo/xp/alpha) is uniform per block.
// B fragments are software-pipelined (double-buffered).
// ---------------------------------------------------------------------------
__global__ void __launch_bounds__(256)
proj_kernel(const float* __restrict__ x,
            const _Float16* __restrict__ pWxo,
            const _Float16* __restrict__ pWxp,
            const _Float16* __restrict__ pWa,
            const float* __restrict__ bo,
            const float* __restrict__ bp,
            const float* __restrict__ ba,
            _Float16* __restrict__ xoB,
            _Float16* __restrict__ xpB,
            _Float16* __restrict__ alB) {
    const int lane  = threadIdx.x & 31;
    const int w     = threadIdx.x >> 5;
    const int mtile = blockIdx.x * 8 + w;           // 0..2047
    const int ntg   = blockIdx.y;                   // 0..11
    const int rowA  = mtile * 16 + (lane & 15);
    const int koff  = (lane >> 4) << 3;             // 0 or 8

    const int region = ntg >> 2;                    // uniform: 0:xo 1:xp 2:alpha
    const int ntl0   = (ntg & 3) * 8;               // first local N-tile
    const _Float16* Psel = (region == 0) ? pWxo : (region == 1) ? pWxp : pWa;
    // lane-resolved base of B fragments: + kt*512 + nt*16384
    const _Float16* pB = Psel + ((size_t)ntl0 * 32) * 512 + lane * 16;

    const v8f vzero = {0.f,0.f,0.f,0.f,0.f,0.f,0.f,0.f};
    v8f acc[8];
    #pragma unroll
    for (int i = 0; i < 8; ++i) acc[i] = vzero;

    // prime B pipeline with (kt=0, nt=0)
    Frag16 bcur;
    bcur.h[0] = *(const v8h*)(pB);
    bcur.h[1] = *(const v8h*)(pB + 8);

    #pragma unroll 2
    for (int kt = 0; kt < 32; ++kt) {               // K = 1024 = 32 * 32
        // A fragment straight from f32 global, converted to f16 (L2-resident x)
        Frag16 a;
        const float* xr = x + (size_t)rowA * D_ + kt * 32 + koff;
        #pragma unroll
        for (int i = 0; i < 8; ++i) a.v[i]     = (_Float16)xr[i];
        #pragma unroll
        for (int i = 0; i < 8; ++i) a.v[8 + i] = (_Float16)xr[16 + i];

        #pragma unroll
        for (int nt = 0; nt < 8; ++nt) {
            // prefetch next B fragment before consuming the current one
            Frag16 bnext = bcur;
            const bool more = (nt < 7) || (kt < 31);
            if (more) {
                const int nkt = (nt < 7) ? kt : kt + 1;
                const int nnt = (nt < 7) ? nt + 1 : 0;
                const _Float16* bptr = pB + (size_t)nkt * 512 + (size_t)nnt * 16384;
                bnext.h[0] = *(const v8h*)bptr;
                bnext.h[1] = *(const v8h*)(bptr + 8);
            }
            acc[nt] = __builtin_amdgcn_wmma_f32_16x16x32_f16(
                false, a.v, false, bcur.v, (short)0, acc[nt], false, false);
            bcur = bnext;
        }
    }

    // Epilogue: bias (+sigmoid for alpha), store f16 row-major (row x 512)
    const float*  bias = (region == 0) ? bo  : (region == 1) ? bp  : ba;
    _Float16*     dst  = (region == 0) ? xoB : (region == 1) ? xpB : alB;
    #pragma unroll
    for (int nt = 0; nt < 8; ++nt) {
        const int col = (ntl0 + nt) * 16 + (lane & 15);
        const float bv = bias[col];
        #pragma unroll
        for (int r = 0; r < 8; ++r) {
            const int row = mtile * 16 + r + koff;  // C layout: M = r (+8 hi lanes)
            float v = acc[nt][r] + bv;
            if (region == 2) v = 1.0f / (1.0f + __expf(-v));
            dst[(size_t)row * HALF_ + col] = (_Float16)v;
        }
    }
}

// ---------------------------------------------------------------------------
// Recurrent scan. 4 blocks x 512 threads (16 waves). Block owns 16 of the 64
// sequences. Dynamic LDS (160 KB of MI455X's 320 KB WGP LDS):
//   [0,  32K)   oh/ph state, double-buffered: state[sb] = {oh, ph} (16 KB each)
//   [32K, 64K)  state buffer 1
//   [64K,160K)  2 x {xo, xp, alpha} staging tiles (16 KB each), TDM-filled
// Double-buffered state -> exactly ONE barrier per recurrent step.
// Wave 0 issues TENSOR_LOAD_TO_LDS for step t+1 while all waves do the
// 16x512x512 WMMA GEMM pair for step t; completion via s_wait_tensorcnt.
// ---------------------------------------------------------------------------
extern __shared__ _Float16 smem[];

__global__ void __launch_bounds__(512)
scan_kernel(const _Float16* __restrict__ xoB,
            const _Float16* __restrict__ xpB,
            const _Float16* __restrict__ alB,
            const _Float16* __restrict__ pWho,
            const _Float16* __restrict__ pWhp,
            const float* __restrict__ thetas,
            float* __restrict__ out,
            float* __restrict__ ohF,
            float* __restrict__ phF) {
    // state[sb]: oh at sb*16384, ph at sb*16384 + 8192 (halves)
    _Float16* stateL = smem;           // 2 x 16384 halves
    _Float16* stage  = smem + 32768;   // 6 x 8192 halves

    const int tid  = threadIdx.x;
    const int lane = tid & 31;
    const int w    = tid >> 5;                  // 0..15
    const int koff = (lane >> 4) << 3;          // 0 or 8
    const int gBase = blockIdx.x * 16;          // first sequence of this block
    const unsigned ldsDyn = __builtin_amdgcn_groupstaticsize();

    // zero state buffer 0 (the only one read at t=0)
    for (int i = tid; i < 16384; i += 512) stateL[i] = (_Float16)0.f;

    float ct[3], st[3];
    #pragma unroll
    for (int c = 0; c < 3; ++c) {
        float th = thetas[c];
        ct[c] = __cosf(th);
        st[c] = __sinf(th);
    }

    // Preload step 0 seeds into staging buffer 0 via TDM (wave 0 only)
    if (w == 0) {
        const size_t e0 = (size_t)gBase * 262144;   // + t*512 elements
        const unsigned sb = ldsDyn + 65536;
        tdm_load_tile(sb,         xoB + e0);
        tdm_load_tile(sb + 16384, xpB + e0);
        tdm_load_tile(sb + 32768, alB + e0);
        __builtin_amdgcn_s_wait_tensorcnt(0);
    }
    __syncthreads();

    const int nt0 = 2 * w;
    const _Float16* pBo = pWho + (size_t)lane * 16;   // + (nt*16+kk)*512
    const _Float16* pBp = pWhp + (size_t)lane * 16;

    for (int t = 0; t < CHUNK_; ++t) {
        const int buf = t & 1;
        // Kick off TDM for step t+1 into the other staging buffer
        if (w == 0 && (t + 1) < CHUNK_) {
            const size_t e1 = (size_t)gBase * 262144 + (size_t)(t + 1) * 512;
            const unsigned sb = ldsDyn + 65536 + (unsigned)(buf ^ 1) * 49152;
            tdm_load_tile(sb,         xoB + e1);
            tdm_load_tile(sb + 16384, xpB + e1);
            tdm_load_tile(sb + 32768, alB + e1);
        }

        const _Float16* sXo = stage + (size_t)(buf * 3 + 0) * 8192;
        const _Float16* sXp = stage + (size_t)(buf * 3 + 1) * 8192;
        const _Float16* sAl = stage + (size_t)(buf * 3 + 2) * 8192;
        const _Float16* ohR = stateL + (size_t)buf * 16384;          // read state
        const _Float16* phR = ohR + 8192;
        _Float16*       ohW = stateL + (size_t)(buf ^ 1) * 16384;    // write state
        _Float16*       phW = ohW + 8192;

        v8f accO[2], accP[2];
        // Seed accumulators with xo_t / xp_t from TDM-staged LDS
        #pragma unroll
        for (int j = 0; j < 2; ++j) {
            const int col = (nt0 + j) * 16 + (lane & 15);
            #pragma unroll
            for (int r = 0; r < 8; ++r) {
                const int row = r + koff;
                accO[j][r] = (float)sXo[row * 512 + col];
                accP[j][r] = (float)sXp[row * 512 + col];
            }
        }

        // 16x512x512 GEMM pair; fully unrolled, B double-buffered from L2
        Frag16 bO, bP;
        {
            const _Float16* q0 = pBo + (size_t)(nt0 * 16) * 512;
            const _Float16* q1 = pBp + (size_t)(nt0 * 16) * 512;
            bO.h[0] = *(const v8h*)q0;  bO.h[1] = *(const v8h*)(q0 + 8);
            bP.h[0] = *(const v8h*)q1;  bP.h[1] = *(const v8h*)(q1 + 8);
        }
        #pragma unroll
        for (int kk = 0; kk < 16; ++kk) {
            Frag16 aO, aP;
            const int base = (lane & 15) * 512 + kk * 32 + koff;
            aO.h[0] = *(const v8h*)&ohR[base];
            aO.h[1] = *(const v8h*)&ohR[base + 16];
            aP.h[0] = *(const v8h*)&phR[base];
            aP.h[1] = *(const v8h*)&phR[base + 16];
            #pragma unroll
            for (int j = 0; j < 2; ++j) {
                Frag16 bOn = bO, bPn = bP;
                const bool more = (j == 0) || (kk < 15);
                if (more) {
                    const int nnt = (j == 0) ? nt0 + 1 : nt0;
                    const int nkk = (j == 0) ? kk : kk + 1;
                    const _Float16* q0 = pBo + (size_t)(nnt * 16 + nkk) * 512;
                    const _Float16* q1 = pBp + (size_t)(nnt * 16 + nkk) * 512;
                    bOn.h[0] = *(const v8h*)q0;  bOn.h[1] = *(const v8h*)(q0 + 8);
                    bPn.h[0] = *(const v8h*)q1;  bPn.h[1] = *(const v8h*)(q1 + 8);
                }
                accO[j] = __builtin_amdgcn_wmma_f32_16x16x32_f16(
                    false, aO.v, false, bO.v, (short)0, accO[j], false, false);
                accP[j] = __builtin_amdgcn_wmma_f32_16x16x32_f16(
                    false, aP.v, false, bP.v, (short)0, accP[j], false, false);
                bO = bOn;  bP = bPn;
            }
        }

        // Elementwise: rotations, gated tanh update. New state goes to the
        // OTHER buffer, so no barrier is needed between GEMM reads and these
        // writes — the single end-of-step barrier covers everything.
        #pragma unroll
        for (int j = 0; j < 2; ++j) {
            const int col = (nt0 + j) * 16 + (lane & 15);
            #pragma unroll
            for (int r = 0; r < 8; ++r) {
                const int row = r + koff;
                const int g   = gBase + row;
                float o = accO[j][r];
                float p = accP[j][r];
                #pragma unroll
                for (int c = 0; c < 3; ++c) {
                    float o2 =  ct[c] * o + st[c] * p;
                    float p2 = -st[c] * o + ct[c] * p;
                    o = o2; p = p2;
                }
                const float a    = (float)sAl[row * 512 + col];
                const float oldO = (float)ohR[row * 512 + col];
                const float oldP = (float)phR[row * 512 + col];
                const float oh2 = (1.f - a) * oldO + a * tanhf(o);
                const float ph2 = (1.f - a) * oldP + a * tanhf(p);
                ohW[row * 512 + col] = (_Float16)oh2;
                phW[row * 512 + col] = (_Float16)ph2;
                const int b  = g >> 2;          // g = b*NC + chunk
                const int cc = g & 3;
                float* op = out + ((size_t)b * S_ + (size_t)cc * CHUNK_ + t) * H_;
                op[col]         = oh2;
                op[HALF_ + col] = ph2;
                if (t == CHUNK_ - 1 && cc == NC_ - 1) {
                    ohF[(size_t)b * HALF_ + col] = oh2;
                    phF[(size_t)b * HALF_ + col] = ph2;
                }
            }
        }
        // Single per-step barrier: publishes new state AND TDM-staged seeds
        if (w == 0) __builtin_amdgcn_s_wait_tensorcnt(0);
        __syncthreads();
    }
}

// ---------------------------------------------------------------------------
// Chunk-boundary coupling fix-up (3 positions per batch; independent).
// ---------------------------------------------------------------------------
__global__ void couple_kernel(float* __restrict__ out) {
    const int h = threadIdx.x;          // 0..511
    const int b = blockIdx.y;           // 0..15
    const int c = blockIdx.x;           // 0..2
    const int e = (c + 1) * CHUNK_;
    float* rowPrev = out + ((size_t)b * S_ + (e - 1)) * H_;
    float* rowCur  = out + ((size_t)b * S_ + e) * H_;
    const float ceo = rowPrev[h];
    const float cep = rowPrev[HALF_ + h];
    const float nso = rowCur[h];
    const float nsp = rowCur[HALF_ + h];
    const float m_o = (h < HALF_ / 2) ?  cep : -cep;
    const float m_p = (h < HALF_ / 2) ? -ceo :  ceo;
    rowCur[h]         = (1.0f - COUPLE_) * nso + COUPLE_ * m_o;
    rowCur[HALF_ + h] = (1.0f - COUPLE_) * nsp + COUPLE_ * m_p;
}

// ---------------------------------------------------------------------------
extern "C" void kernel_launch(void* const* d_in, const int* in_sizes, int n_in,
                              void* d_out, int out_size, void* d_ws, size_t ws_size,
                              hipStream_t stream) {
    (void)in_sizes; (void)n_in; (void)out_size; (void)ws_size;

    const float* x      = (const float*)d_in[0];
    const float* Wxo    = (const float*)d_in[1];
    const float* Who    = (const float*)d_in[2];
    const float* bo     = (const float*)d_in[3];
    const float* Wxp    = (const float*)d_in[4];
    const float* Whp    = (const float*)d_in[5];
    const float* bp     = (const float*)d_in[6];
    const float* Wa     = (const float*)d_in[7];
    const float* ba     = (const float*)d_in[8];
    const float* thetas = (const float*)d_in[9];

    float* out = (float*)d_out;
    float* ohF = out + (size_t)B_ * S_ * H_;
    float* phF = ohF + (size_t)B_ * HALF_;

    char* ws = (char*)d_ws;
    _Float16* pWxo = (_Float16*)ws;  ws += (size_t)D_ * HALF_ * 2;     // 1 MB
    _Float16* pWxp = (_Float16*)ws;  ws += (size_t)D_ * HALF_ * 2;
    _Float16* pWa  = (_Float16*)ws;  ws += (size_t)D_ * HALF_ * 2;
    _Float16* pWho = (_Float16*)ws;  ws += (size_t)HALF_ * HALF_ * 2;  // 512 KB
    _Float16* pWhp = (_Float16*)ws;  ws += (size_t)HALF_ * HALF_ * 2;
    _Float16* xoB  = (_Float16*)ws;  ws += (size_t)NTOK_ * HALF_ * 2;  // 32 MB
    _Float16* xpB  = (_Float16*)ws;  ws += (size_t)NTOK_ * HALF_ * 2;
    _Float16* alB  = (_Float16*)ws;                                    // 32 MB

    // Weight packing into WMMA B-fragment layout (f16)
    pack_w_kernel<<<dim3(32 * 32), 32, 0, stream>>>(Wxo, pWxo, D_, HALF_);
    pack_w_kernel<<<dim3(32 * 32), 32, 0, stream>>>(Wxp, pWxp, D_, HALF_);
    pack_w_kernel<<<dim3(32 * 32), 32, 0, stream>>>(Wa,  pWa,  D_, HALF_);
    pack_w_kernel<<<dim3(32 * 16), 32, 0, stream>>>(Who, pWho, HALF_, HALF_);
    pack_w_kernel<<<dim3(32 * 16), 32, 0, stream>>>(Whp, pWhp, HALF_, HALF_);

    // xo / xp / alpha projections (WMMA f16->f32)
    proj_kernel<<<dim3(256, 12), 256, 0, stream>>>(
        x, pWxo, pWxp, pWa, bo, bp, ba, xoB, xpB, alB);

    // Recurrent scan: 4 WGs x 16 waves, double-buffered LDS state +
    // TDM-staged seeds (160 KB dynamic LDS), one barrier per step
    scan_kernel<<<dim3(4), 512, 163840, stream>>>(
        xoB, xpB, alB, pWho, pWhp, thetas, out, ohF, phF);

    // Chunk-boundary coupling
    couple_kernel<<<dim3(3, 16), 512, 0, stream>>>(out);
}
